// LocalFeatureTransformer_31722628448330
// MI455X (gfx1250) — matrix operand order repaired and checked by reading, not compiled, not run on hardware
//
#include <hip/hip_runtime.h>
#include <hip/hip_bf16.h>
#include <cstddef>

// ---------------------------------------------------------------------------
// Problem constants (match reference)
// ---------------------------------------------------------------------------
#define CDIM   256
#define NH     8
#define HD     32
#define NPROTO 8
#define BB     8
#define LTOK   4800
#define NTOK   (BB * LTOK)   // 38400 rows

enum { MODE_PLAIN = 0, MODE_ELU1 = 1, MODE_RELU = 2 };

typedef __bf16 bf16_t;
typedef __attribute__((ext_vector_type(16))) __bf16 v16bf;
typedef __attribute__((ext_vector_type(8)))  __bf16 bf16x8;
typedef __attribute__((ext_vector_type(8)))  float  v8f;

// ---------------------------------------------------------------------------
// WMMA GEMM: Y[M,Nc] = act( A[M,K] @ W[K,Nc] ), A bf16 [M,K] row-major,
// W pre-transposed bf16 WT[Nc,K]: every fragment = two contiguous
// global_load_b128. fp32 accumulate via v_wmma_f32_16x16x32_bf16.
// Software-pipelined k-loop (register double buffer): fragments for kb+32 are
// issued before the 4 WMMAs of kb execute, hiding load latency.
// Split-A feeds concat([x,msg]) without materializing it (uniform scalar
// select on kb, hoisted row pointers).
// Block = 256 threads = 8 waves; block tile 128x64; wave tile 16x64.
// ---------------------------------------------------------------------------
__global__ __launch_bounds__(256)
void gemm_wmma_kernel(const bf16_t* __restrict__ A0, const bf16_t* __restrict__ A1,
                      int splitK, int lda0, int lda1,
                      const bf16_t* __restrict__ WT,   // [Nc, K]
                      float* __restrict__ Yf, bf16_t* __restrict__ Yb,
                      int M, int K, int Nc, int mode)
{
    const int wave = threadIdx.x >> 5;
    const int lane = threadIdx.x & 31;
    const int hsel = lane >> 4;    // 0 or 1
    const int ln16 = lane & 15;
    const int row0 = blockIdx.x * 128 + wave * 16;
    const int col0 = blockIdx.y * 64;
    if (row0 >= M) return;

    const int arow = row0 + ln16;                          // A row this lane reads
    const bf16_t* const Arow0 = A0 + (size_t)arow * lda0 + hsel * 8;
    const bf16_t* const Arow1 = A1 + (size_t)arow * lda1 + hsel * 8;
    const bf16_t* const Bbase = WT + (size_t)(col0 + ln16) * K + hsel * 8;

    // A fragment (16-bit A 16x32): lanes 0-15 K in {0..7}U{16..23}; lanes 16-31 +8
    auto loadA = [&](int kb) -> v16bf {
        const bf16_t* p = (kb < splitK) ? (Arow0 + kb) : (Arow1 + (kb - splitK));
        const bf16x8 lo = *(const bf16x8*)p;
        const bf16x8 hi = *(const bf16x8*)(p + 16);
        return __builtin_shufflevector(lo, hi, 0,1,2,3,4,5,6,7,8,9,10,11,12,13,14,15);
    };
    // B fragment (32x16): lanes 0-15 K in {0..7}U{16..23}; lanes 16-31 +8 (matches A)
    auto loadB = [&](int kb, int t) -> v16bf {
        const bf16_t* p = Bbase + (size_t)t * 16 * K + kb;
        const bf16x8 lo = *(const bf16x8*)p;
        const bf16x8 hi = *(const bf16x8*)(p + 16);
        return __builtin_shufflevector(lo, hi, 0,1,2,3,4,5,6,7,8,9,10,11,12,13,14,15);
    };

    v8f acc[4] = {};

    // prologue: fragments for kb = 0
    v16bf aC  = loadA(0);
    v16bf bC0 = loadB(0, 0), bC1 = loadB(0, 1), bC2 = loadB(0, 2), bC3 = loadB(0, 3);

    for (int kb = 0; kb < K; kb += 32) {
        v16bf aN = aC, bN0 = bC0, bN1 = bC1, bN2 = bC2, bN3 = bC3;
        const int kn = kb + 32;
        if (kn < K) {   // issue next-step loads BEFORE consuming current fragments
            aN  = loadA(kn);
            bN0 = loadB(kn, 0); bN1 = loadB(kn, 1);
            bN2 = loadB(kn, 2); bN3 = loadB(kn, 3);
        }
        acc[0] = __builtin_amdgcn_wmma_f32_16x16x32_bf16(false, aC, false, bC0, (short)0, acc[0], false, false);
        acc[1] = __builtin_amdgcn_wmma_f32_16x16x32_bf16(false, aC, false, bC1, (short)0, acc[1], false, false);
        acc[2] = __builtin_amdgcn_wmma_f32_16x16x32_bf16(false, aC, false, bC2, (short)0, acc[2], false, false);
        acc[3] = __builtin_amdgcn_wmma_f32_16x16x32_bf16(false, aC, false, bC3, (short)0, acc[3], false, false);
        aC = aN; bC0 = bN0; bC1 = bN1; bC2 = bN2; bC3 = bN3;
    }

    // C/D layout: VGPR r <-> M = r + 8*hsel, N = lane&15
#pragma unroll
    for (int t = 0; t < 4; ++t) {
        const int col = col0 + t * 16 + ln16;
#pragma unroll
        for (int r = 0; r < 8; ++r) {
            const int row = row0 + r + hsel * 8;
            float y = acc[t][r];
            if (mode == MODE_ELU1)      y = (y > 0.f) ? (y + 1.f) : __expf(y); // elu(y)+1
            else if (mode == MODE_RELU) y = (y > 0.f) ? y : 0.f;
            if (row < M) {
                if (Yf) Yf[(size_t)row * Nc + col] = y;
                if (Yb) Yb[(size_t)row * Nc + col] = (bf16_t)y;
            }
        }
    }
}

// ---------------------------------------------------------------------------
// Prototype classification: f_p[r,p] = x[r,:] . proto[p,:], argmax -> class.
// ---------------------------------------------------------------------------
__global__ __launch_bounds__(256)
void classify_kernel(const float* __restrict__ x, const float* __restrict__ proto,
                     int* __restrict__ cls, float* __restrict__ fp_out,
                     float* __restrict__ cls_out, int M)
{
    const int r = blockIdx.x * 256 + threadIdx.x;
    if (r >= M) return;
    const float* xr = x + (size_t)r * CDIM;
    float best = -3.4e38f; int bi = 0;
#pragma unroll
    for (int p = 0; p < NPROTO; ++p) {
        float s = 0.f;
        for (int c = 0; c < CDIM; ++c) s += xr[c] * proto[p * CDIM + c];
        fp_out[(size_t)r * NPROTO + p] = s;
        if (s > best) { best = s; bi = p; }   // strict > : first max wins (argmax)
    }
    cls[r] = bi;
    cls_out[r] = (float)bi;
}

// ---------------------------------------------------------------------------
// kv[b,p,h,d,e] += sum_s oh[s,p] * phi_k[s,h,d] * v[s,h,e]
// ksum[b,p,h,d] += sum_s oh[s,p] * phi_k[s,h,d]
// Grid: (B*H, SCHUNKS); LDS-accumulated per class, atomic flush to global.
// ---------------------------------------------------------------------------
#define SCHUNKS 8
#define SCHUNK  (LTOK / SCHUNKS)   // 600
__global__ __launch_bounds__(256)
void kv_accum_kernel(const float* __restrict__ phik, const float* __restrict__ v,
                     const int* __restrict__ cls, const int* __restrict__ valid,
                     float* __restrict__ kvg, float* __restrict__ ksg)
{
    __shared__ float acc[NPROTO * HD * HD];  // 32 KB
    __shared__ float ks[NPROTO * HD];        //  1 KB
    const int t  = threadIdx.x;
    const int b  = blockIdx.x >> 3;
    const int h  = blockIdx.x & 7;
    const int e  = t & 31;
    const int d0 = t >> 5;   // 0..7

    for (int i = t; i < NPROTO * HD * HD; i += 256) acc[i] = 0.f;
    if (t < NPROTO * HD) ks[t] = 0.f;
    __syncthreads();

    const int s0 = blockIdx.y * SCHUNK;
    for (int s = s0; s < s0 + SCHUNK; ++s) {
        const int row = b * LTOK + s;
        if (!valid[row]) continue;
        const int p = cls[row];
        const float vv = v[(size_t)row * CDIM + h * HD + e];
        float* ap = acc + p * HD * HD;
#pragma unroll
        for (int j = 0; j < 4; ++j) {
            const int d = d0 + j * 8;
            const float pk = phik[(size_t)row * CDIM + h * HD + d];
            ap[d * HD + e] += pk * vv;
        }
        if (t < HD) ks[p * HD + t] += phik[(size_t)row * CDIM + h * HD + t];
    }
    __syncthreads();

#pragma unroll
    for (int p = 0; p < NPROTO; ++p)
#pragma unroll
        for (int j = 0; j < 4; ++j) {
            const int q = t + j * 256;
            atomicAdd(&kvg[((((size_t)b * NPROTO + p) * NH + h) << 10) + q],
                      acc[p * HD * HD + q]);
        }
    {   // 256 entries of ksum: p = t>>5, d = t&31
        const int p = t >> 5, d = t & 31;
        atomicAdd(&ksg[((((size_t)b * NPROTO + p) * NH + h) << 5) + d], ks[t]);
    }
}

// ---------------------------------------------------------------------------
// msg[r, h*32+e] = (phi_q[r,h,:] . kv[b,p,h,:,e]) / (phi_q . ksum + eps)
// One wave per (token,head): lane = e. Output written as bf16 (GEMM input).
// ---------------------------------------------------------------------------
__global__ __launch_bounds__(256)
void msg_kernel(const float* __restrict__ phiq, const float* __restrict__ kvg,
                const float* __restrict__ ksg, const int* __restrict__ cls,
                const int* __restrict__ valid, bf16_t* __restrict__ msgb)
{
    const int unit = blockIdx.x * 8 + (threadIdx.x >> 5); // r*NH + h
    const int lane = threadIdx.x & 31;
    const int h = unit & 7;
    const int r = unit >> 3;
    if (r >= NTOK) return;
    bf16_t* outp = msgb + (size_t)r * CDIM + h * HD + lane;
    if (!valid[r]) { *outp = (bf16_t)0.f; return; }
    const int b = r / LTOK;
    const int p = cls[r];
    const float* pq = phiq + (size_t)r * CDIM + h * HD;
    const float* KV = kvg + ((((size_t)b * NPROTO + p) * NH + h) << 10);
    const float* KS = ksg + ((((size_t)b * NPROTO + p) * NH + h) << 5);
    float num = 0.f, den = 0.f;
#pragma unroll 8
    for (int d = 0; d < HD; ++d) {
        const float q = pq[d];
        num += q * KV[d * HD + lane];
        den += q * KS[d];
    }
    *outp = (bf16_t)(num / (den + 1e-6f));
}

// ---------------------------------------------------------------------------
// In-place LayerNorm over C=256 with affine; optional bf16 shadow output.
// ---------------------------------------------------------------------------
__global__ __launch_bounds__(256)
void ln_kernel(float* __restrict__ X, const float* __restrict__ g,
               const float* __restrict__ bt, bf16_t* __restrict__ outb)
{
    __shared__ float red[256];
    const int row = blockIdx.x;
    const int c = threadIdx.x;
    float x = X[(size_t)row * CDIM + c];
    red[c] = x; __syncthreads();
    for (int off = 128; off > 0; off >>= 1) {
        if (c < off) red[c] += red[c + off];
        __syncthreads();
    }
    const float mean = red[0] * (1.f / 256.f);
    __syncthreads();
    const float d = x - mean;
    red[c] = d * d; __syncthreads();
    for (int off = 128; off > 0; off >>= 1) {
        if (c < off) red[c] += red[c + off];
        __syncthreads();
    }
    const float var = red[0] * (1.f / 256.f);
    const float y = d * rsqrtf(var + 1e-5f) * g[c] + bt[c];
    X[(size_t)row * CDIM + c] = y;
    if (outb) outb[(size_t)row * CDIM + c] = (bf16_t)y;
}

// dst = valid ? x + m : x   (one block per row); also writes bf16 shadow
__global__ __launch_bounds__(256)
void residual_mask_kernel(const float* __restrict__ x, const float* __restrict__ m,
                          const int* __restrict__ valid,
                          float* __restrict__ dst, bf16_t* __restrict__ dstb)
{
    const int r = blockIdx.x;
    const size_t i = (size_t)r * CDIM + threadIdx.x;
    const float y = valid[r] ? (x[i] + m[i]) : x[i];
    dst[i] = y;
    dstb[i] = (bf16_t)y;
}

__global__ void fill_zero_kernel(float* __restrict__ p, size_t n)
{
    const size_t i = (size_t)blockIdx.x * 256 + threadIdx.x;
    if (i < n) p[i] = 0.f;
}

__global__ void copy_f_kernel(const float* __restrict__ s, float* __restrict__ d, size_t n)
{
    const size_t i = (size_t)blockIdx.x * 256 + threadIdx.x;
    if (i < n) d[i] = s[i];
}

__global__ void cvt_bf16_kernel(const float* __restrict__ s, bf16_t* __restrict__ d, size_t n)
{
    const size_t i = (size_t)blockIdx.x * 256 + threadIdx.x;
    if (i < n) d[i] = (bf16_t)s[i];
}

// WT[n*K + k] = (bf16) W[k*Nc + n]   (weights: tiny, done once per launch)
__global__ void wtrans_kernel(const float* __restrict__ Wsrc, bf16_t* __restrict__ WT,
                              int K, int Nc)
{
    const int idx = blockIdx.x * 256 + threadIdx.x;
    if (idx >= K * Nc) return;
    const int n = idx / K, k = idx - n * K;
    WT[(size_t)n * K + k] = (bf16_t)Wsrc[(size_t)k * Nc + n];
}

// ---------------------------------------------------------------------------
// Host orchestration
// ---------------------------------------------------------------------------
extern "C" void kernel_launch(void* const* d_in, const int* in_sizes, int n_in,
                              void* d_out, int out_size, void* d_ws, size_t ws_size,
                              hipStream_t stream)
{
    const float* feat0 = (const float*)d_in[0];
    const float* feat1 = (const float*)d_in[1];
    const int*   mask0 = (const int*)d_in[2];
    const int*   mask1 = (const int*)d_in[3];
    const float* f0wo  = (const float*)d_in[4];
    const float* f1wo  = (const float*)d_in[5];
    const float* proto = (const float*)d_in[6];
    const float* Wq    = (const float*)d_in[7];
    const float* Wk    = (const float*)d_in[8];
    const float* Wv    = (const float*)d_in[9];
    const float* Wm    = (const float*)d_in[10];
    const float* W1    = (const float*)d_in[11];
    const float* W2    = (const float*)d_in[12];
    const float* g1    = (const float*)d_in[13];
    const float* b1    = (const float*)d_in[14];
    const float* g2    = (const float*)d_in[15];
    const float* b2    = (const float*)d_in[16];

    // ---------------- workspace layout ----------------
    float* w = (float*)d_ws;
    size_t o = 0;
    float* P0   = w + o; o += (size_t)NTOK * CDIM;   // phi_q / MLP out
    float* P1   = w + o; o += (size_t)NTOK * CDIM;   // phi_k
    float* P2   = w + o; o += (size_t)NTOK * CDIM;   // v
    float* P4   = w + o; o += (size_t)NTOK * CDIM;   // msg @ Wm (LN in place)
    float* f0c  = w + o; o += (size_t)NTOK * CDIM;
    float* f1c  = w + o; o += (size_t)NTOK * CDIM;
    float* new0 = w + o; o += (size_t)NTOK * CDIM;
    float* kvg  = w + o; o += (size_t)BB * NPROTO * NH * HD * HD;  // 524288
    float* ksg  = w + o; o += (size_t)BB * NPROTO * NH * HD;       // 16384
    int*   cls0 = (int*)(w + o); o += NTOK;
    int*   cls1 = (int*)(w + o); o += NTOK;

    bf16_t* bb = (bf16_t*)(w + o);
    size_t ob = 0;
    bf16_t* f0cb  = bb + ob; ob += (size_t)NTOK * CDIM;
    bf16_t* f1cb  = bb + ob; ob += (size_t)NTOK * CDIM;
    bf16_t* new0b = bb + ob; ob += (size_t)NTOK * CDIM;
    bf16_t* P3b   = bb + ob; ob += (size_t)NTOK * CDIM;      // msg (bf16)
    bf16_t* P4b   = bb + ob; ob += (size_t)NTOK * CDIM;      // msg_ln (bf16)
    bf16_t* P5b   = bb + ob; ob += (size_t)NTOK * 2 * CDIM;  // MLP hidden (bf16)
    bf16_t* WTq   = bb + ob; ob += 4 * (size_t)CDIM * CDIM;
    bf16_t* WTk   = bb + ob; ob += 4 * (size_t)CDIM * CDIM;
    bf16_t* WTv   = bb + ob; ob += 4 * (size_t)CDIM * CDIM;
    bf16_t* WTm   = bb + ob; ob += 4 * (size_t)CDIM * CDIM;
    bf16_t* WT1   = bb + ob; ob += 4 * (size_t)(2 * CDIM) * (2 * CDIM);
    bf16_t* WT2   = bb + ob; ob += 4 * (size_t)(2 * CDIM) * CDIM;

    // ---------------- output layout (flat, reference return order) ----------
    float* out     = (float*)d_out;
    float* out_f0  = out;
    float* out_f1  = out_f0 + (size_t)NTOK * CDIM;
    float* out_c0  = out_f1 + (size_t)NTOK * CDIM;
    float* out_c1  = out_c0 + NTOK;
    float* out_fp0 = out_c1 + NTOK;
    float* out_fp1 = out_fp0 + (size_t)NTOK * NPROTO;
    float* out_pr  = out_fp1 + (size_t)NTOK * NPROTO;

    const size_t featBytes  = (size_t)NTOK * CDIM * sizeof(float);
    const size_t featBytesB = (size_t)NTOK * CDIM * sizeof(bf16_t);

    auto gemm = [&](const bf16_t* A0, const bf16_t* A1, int splitK, int lda0, int lda1,
                    const bf16_t* WTp, float* Yf, bf16_t* Yb,
                    int M, int K, int Nc, int mode) {
        dim3 grid((M + 127) / 128, Nc / 64);
        gemm_wmma_kernel<<<grid, 256, 0, stream>>>(A0, A1, splitK, lda0, lda1,
                                                   WTp, Yf, Yb, M, K, Nc, mode);
    };

    auto grouped = [&](const float* xf, const bf16_t* xb, const bf16_t* srcb,
                       const int* clsx, const int* vx,
                       const int* clss, const int* vs,
                       int layer, float* dstf, bf16_t* dstb) {
        const bf16_t* wq = WTq + (size_t)layer * CDIM * CDIM;
        const bf16_t* wk = WTk + (size_t)layer * CDIM * CDIM;
        const bf16_t* wv = WTv + (size_t)layer * CDIM * CDIM;
        const bf16_t* wm = WTm + (size_t)layer * CDIM * CDIM;
        const bf16_t* w1 = WT1 + (size_t)layer * 2 * CDIM * 2 * CDIM;
        const bf16_t* w2 = WT2 + (size_t)layer * 2 * CDIM * CDIM;

        // phi_q = elu(x@Wq)+1 ; phi_k = elu(src@Wk)+1 ; v = src@Wv
        gemm(xb,   xb,   CDIM, CDIM, CDIM, wq, P0, nullptr, NTOK, CDIM, CDIM, MODE_ELU1);
        gemm(srcb, srcb, CDIM, CDIM, CDIM, wk, P1, nullptr, NTOK, CDIM, CDIM, MODE_ELU1);
        gemm(srcb, srcb, CDIM, CDIM, CDIM, wv, P2, nullptr, NTOK, CDIM, CDIM, MODE_PLAIN);

        // per-class linear attention statistics (kv + ksum are contiguous)
        const size_t kvTot = (size_t)BB * NPROTO * NH * HD * HD
                           + (size_t)BB * NPROTO * NH * HD;
        fill_zero_kernel<<<(unsigned)((kvTot + 255) / 256), 256, 0, stream>>>(kvg, kvTot);
        kv_accum_kernel<<<dim3(BB * NH, SCHUNKS), 256, 0, stream>>>(P1, P2, clss, vs, kvg, ksg);

        // msg = num / den  (bf16, feeds Wm GEMM)
        msg_kernel<<<NTOK, 256, 0, stream>>>(P0, kvg, ksg, clsx, vx, P3b);

        // msg = LN(msg @ Wm) * g1 + b1  (fp32 in P4, bf16 shadow in P4b)
        gemm(P3b, P3b, CDIM, CDIM, CDIM, wm, P4, nullptr, NTOK, CDIM, CDIM, MODE_PLAIN);
        ln_kernel<<<NTOK, 256, 0, stream>>>(P4, g1 + layer * CDIM, b1 + layer * CDIM, P4b);

        // MLP: h = relu(concat([x, msg]) @ W1) (bf16) ; m = LN(h @ W2) * g2 + b2
        gemm(xb,  P4b, CDIM, CDIM, CDIM, w1, nullptr, P5b, NTOK, 2 * CDIM, 2 * CDIM, MODE_RELU);
        gemm(P5b, P5b, 2 * CDIM, 2 * CDIM, 2 * CDIM, w2, P0, nullptr, NTOK, 2 * CDIM, CDIM, MODE_PLAIN);
        ln_kernel<<<NTOK, 256, 0, stream>>>(P0, g2 + layer * CDIM, b2 + layer * CDIM, nullptr);

        // dst = valid ? x + m : x  (fp32 + bf16 shadow)
        residual_mask_kernel<<<NTOK, 256, 0, stream>>>(xf, P0, vx, dstf, dstb);
    };

    // ---- weights: convert + transpose to bf16 [Nc,K] (tiny, once) ----
    for (int l = 0; l < 4; ++l) {
        const int E = CDIM * CDIM;
        wtrans_kernel<<<(E + 255) / 256, 256, 0, stream>>>(Wq + (size_t)l * E, WTq + (size_t)l * E, CDIM, CDIM);
        wtrans_kernel<<<(E + 255) / 256, 256, 0, stream>>>(Wk + (size_t)l * E, WTk + (size_t)l * E, CDIM, CDIM);
        wtrans_kernel<<<(E + 255) / 256, 256, 0, stream>>>(Wv + (size_t)l * E, WTv + (size_t)l * E, CDIM, CDIM);
        wtrans_kernel<<<(E + 255) / 256, 256, 0, stream>>>(Wm + (size_t)l * E, WTm + (size_t)l * E, CDIM, CDIM);
        const int E1 = 2 * CDIM * 2 * CDIM;
        wtrans_kernel<<<(E1 + 255) / 256, 256, 0, stream>>>(W1 + (size_t)l * E1, WT1 + (size_t)l * E1, 2 * CDIM, 2 * CDIM);
        const int E2 = 2 * CDIM * CDIM;
        wtrans_kernel<<<(E2 + 255) / 256, 256, 0, stream>>>(W2 + (size_t)l * E2, WT2 + (size_t)l * E2, 2 * CDIM, CDIM);
    }

    // ---- classification + static outputs ----
    classify_kernel<<<(NTOK + 255) / 256, 256, 0, stream>>>(f0wo, proto, cls0, out_fp0, out_c0, NTOK);
    classify_kernel<<<(NTOK + 255) / 256, 256, 0, stream>>>(f1wo, proto, cls1, out_fp1, out_c1, NTOK);
    copy_f_kernel<<<(NPROTO * CDIM + 255) / 256, 256, 0, stream>>>(proto, out_pr, NPROTO * CDIM);

    hipMemcpyAsync(f0c, feat0, featBytes, hipMemcpyDeviceToDevice, stream);
    hipMemcpyAsync(f1c, feat1, featBytes, hipMemcpyDeviceToDevice, stream);
    cvt_bf16_kernel<<<(NTOK * CDIM + 255) / 256, 256, 0, stream>>>(feat0, f0cb, (size_t)NTOK * CDIM);
    cvt_bf16_kernel<<<(NTOK * CDIM + 255) / 256, 256, 0, stream>>>(feat1, f1cb, (size_t)NTOK * CDIM);

    // ---- 4 layers: self-self, cross-self, self-self, cross-self ----
    for (int i = 0; i < 4; ++i) {
        if ((i & 1) == 0) {   // self-self
            grouped(f0c, f0cb, f0cb, cls0, mask0, cls0, mask0, i, f0c, f0cb);
            grouped(f1c, f1cb, f1cb, cls1, mask1, cls1, mask1, i, f1c, f1cb);
        } else {              // cross-self (feat1 attends UPDATED feat0)
            grouped(f0c, f0cb, f1cb,  cls0, mask0, cls1, mask1, i, new0, new0b);
            grouped(f1c, f1cb, new0b, cls1, mask1, cls0, mask0, i, f1c, f1cb);
            hipMemcpyAsync(f0c,  new0,  featBytes,  hipMemcpyDeviceToDevice, stream);
            hipMemcpyAsync(f0cb, new0b, featBytesB, hipMemcpyDeviceToDevice, stream);
        }
    }

    hipMemcpyAsync(out_f0, f0c, featBytes, hipMemcpyDeviceToDevice, stream);
    hipMemcpyAsync(out_f1, f1c, featBytes, hipMemcpyDeviceToDevice, stream);
}
